// WaveletKANLayer_26233660244248
// MI455X (gfx1250) — matrix operand order repaired and checked by reading, not compiled, not run on hardware
//
#include <hip/hip_runtime.h>
#include <hip/hip_bf16.h>

// ---------------------------------------------------------------------------
// WaveletKAN layer for MI455X (gfx1250, wave32, WMMA)
//   Phase 1: 4-level db4 DWT per row (LDS-resident), emit bf16 hi/lo coeffs
//   Phase 2: split-precision bf16x3 WMMA GEMM, double-buffered LDS,
//            async global->LDS copies for A tiles, AS1 global loads for B.
// ---------------------------------------------------------------------------

typedef __attribute__((ext_vector_type(16))) __bf16 v16bf;
typedef __attribute__((ext_vector_type(8)))  float  v8f;
typedef __attribute__((ext_vector_type(4)))  int    v4i;
typedef __attribute__((ext_vector_type(4)))  float  v4f;

typedef const __attribute__((address_space(1))) float cgfloat;
typedef const __attribute__((address_space(1))) v4f   cgv4f;

#define KTOT 4256   // padded total K: 288+288+544+1056+2080

#if defined(__has_builtin)
#if __has_builtin(__builtin_amdgcn_global_load_async_to_lds_b128) && \
    __has_builtin(__builtin_amdgcn_s_wait_asynccnt)
#define USE_ASYNC_COPY 1
#endif
#endif

#ifdef USE_ASYNC_COPY
// builtin signature (per hipcc diagnostic): (v4i AS1*, v4i AS3*, imm int, imm int)
#define GAS1(p) ((__attribute__((address_space(1))) v4i*)(p))
#define LAS3(p) ((__attribute__((address_space(3))) v4i*)(p))
#endif

__device__ __forceinline__ unsigned short f2bf(float f) {
    unsigned u = __float_as_uint(f);
    u += 0x7FFFu + ((u >> 16) & 1u);     // round-to-nearest-even
    return (unsigned short)(u >> 16);
}
__device__ __forceinline__ float bf2f(unsigned short h) {
    return __uint_as_float(((unsigned)h) << 16);
}

// ---------------------------------------------------------------------------
// Phase 1: db4 wavedec, 4 levels, symmetric extension.
// out[m] = sum_u filt[u] * a[sym(2m+1-u)],  M = (N+7)/2
// Coeff layout in ws (per row, bf16 hi & lo planes):
//   [cA4 @0 pad288][cD4 @288 pad288][cD3 @576 pad544][cD2 @1120 pad1056][cD1 @2176 pad2080]
// ---------------------------------------------------------------------------
__global__ __launch_bounds__(256)
void dwt_db4_kernel(const float* __restrict__ x,
                    unsigned short* __restrict__ Ahi,
                    unsigned short* __restrict__ Alo)
{
    __shared__ __attribute__((aligned(16))) float bufA[4096];
    __shared__ __attribute__((aligned(16))) float bufB[2052];

    const int row = blockIdx.x;
    const int tid = threadIdx.x;

    const float* xr = x + (size_t)row * 4096;
    for (int i = tid; i < 4096; i += 256) bufA[i] = xr[i];
    __syncthreads();

    const float dlo[8] = {-0.010597401784997278f,  0.032883011666982945f,
                           0.030841381835986965f, -0.18703481171888114f,
                          -0.02798376941698385f,   0.6308807679295904f,
                           0.7148465705525415f,    0.23037781330885523f};
    const float dhi[8] = {-0.23037781330885523f,   0.7148465705525415f,
                          -0.6308807679295904f,   -0.02798376941698385f,
                           0.18703481171888114f,   0.030841381835986965f,
                          -0.032883011666982945f, -0.010597401784997278f};

    const int Nlev[4] = {4096, 2051, 1029, 518};
    const int doff[4] = {2176, 1120, 576, 288};   // cD1, cD2, cD3, cD4

    unsigned short* hrow = Ahi + (size_t)row * KTOT;
    unsigned short* lrow = Alo + (size_t)row * KTOT;

    float* cur = bufA;
    float* nxt = bufB;
    for (int lev = 0; lev < 4; ++lev) {
        const int N = Nlev[lev];
        const int M = (N + 7) >> 1;
        for (int m = tid; m < M; m += 256) {
            float sA = 0.f, sD = 0.f;
            #pragma unroll
            for (int u = 0; u < 8; ++u) {
                int idx = 2 * m + 1 - u;
                idx = (idx < 0) ? (-1 - idx) : idx;
                idx = (idx >= N) ? (2 * N - 1 - idx) : idx;
                float v = cur[idx];
                sA += dlo[u] * v;
                sD += dhi[u] * v;
            }
            nxt[m] = sA;
            const int c = doff[lev] + m;
            unsigned short hd = f2bf(sD);
            hrow[c] = hd;
            lrow[c] = f2bf(sD - bf2f(hd));
            if (lev == 3) {                      // final approximation cA4 @0
                unsigned short ha = f2bf(sA);
                hrow[m] = ha;
                lrow[m] = f2bf(sA - bf2f(ha));
            }
        }
        __syncthreads();
        float* t = cur; cur = nxt; nxt = t;
    }

    // zero the per-segment K padding so the GEMM can run full 32-wide steps
    const int ps[5] = {262, 550, 1094, 2149, 4227};
    const int pe[5] = {288, 576, 1120, 2176, 4256};
    #pragma unroll
    for (int s = 0; s < 5; ++s)
        for (int c = ps[s] + tid; c < pe[s]; c += 256) {
            hrow[c] = 0; lrow[c] = 0;
        }
}

// ---------------------------------------------------------------------------
// Phase 2: C[4096,4096] = sum_s Acoef_s * Basis_s, bf16x3 split precision.
// Workgroup: 256 threads = 8 waves, tile 128(M) x 128(N), double-buffered LDS.
// K is flattened into 133 uniform 32-wide steps across the 5 segments.
// Pipeline per step: [wait async; barrier] -> issue loads for t+1 (async A +
// B f32 into regs) -> 24 WMMAs on tile t -> bf16-split + ds_store of B(t+1).
// ---------------------------------------------------------------------------
union FragU {
    uint4 q[2];
    v16bf v;
};

__global__ __launch_bounds__(256)
void wkan_wmma_gemm(const unsigned short* __restrict__ Ahi,
                    const unsigned short* __restrict__ Alo,
                    const float* __restrict__ b0, const float* __restrict__ b1,
                    const float* __restrict__ b2, const float* __restrict__ b3,
                    const float* __restrict__ b4,
                    float* __restrict__ out)
{
    constexpr int LDA = 40;            // LDS row stride A (32 + 8 pad) bf16 elems
    constexpr int LDB = 136;           // LDS row stride B (128 + 8 pad) bf16 elems
    constexpr int ASZ = 128 * LDA;     // one A buffer (elems)
    constexpr int BSZ = 32 * LDB;      // one B buffer (elems)
    constexpr int NSTEP = 133;         // total 32-wide K steps

    __shared__ __attribute__((aligned(16))) unsigned short AsH[2 * ASZ];
    __shared__ __attribute__((aligned(16))) unsigned short AsL[2 * ASZ];
    __shared__ __attribute__((aligned(16))) unsigned short BsH[2 * BSZ];
    __shared__ __attribute__((aligned(16))) unsigned short BsL[2 * BSZ];

    const int tid   = threadIdx.x;
    const int lane  = tid & 31;
    const int wave  = tid >> 5;
    const int waveM = wave >> 2;       // 0..1
    const int waveN = wave & 3;        // 0..3
    const int m0    = blockIdx.y * 128;
    const int n0    = blockIdx.x * 128;

    // keep basis pointers in the global address space so B loads stay
    // global_load_b128 (not flat_load) even after the dynamic segment select
    cgfloat* bases[5] = {(cgfloat*)b0, (cgfloat*)b1, (cgfloat*)b2,
                         (cgfloat*)b3, (cgfloat*)b4};
    const int aoff[5] = {0, 288, 576, 1120, 2176};
    const int scum[6] = {0, 9, 18, 35, 68, 133};   // cumulative K-steps/segment

    const int lh = lane >> 4;          // K-half select per ISA A/B layouts
    const int lm = lane & 15;

    // per-thread staging chunk coordinates (fixed for all steps)
    int ar[2], akc[2];                 // A: 512 chunks of 8 bf16
    #pragma unroll
    for (int i = 0; i < 2; ++i) {
        int chunk = tid + 256 * i;
        ar[i]  = chunk >> 2;
        akc[i] = (chunk & 3) * 8;
    }
    int bk[4], bcg[4];                 // B: 1024 chunks of 4 f32
    #pragma unroll
    for (int i = 0; i < 4; ++i) {
        int chunk = tid + 256 * i;
        bk[i]  = chunk >> 5;
        bcg[i] = (chunk & 31) * 4;
    }

    v8f acc[4][2] = {};
    v4f bReg[4];
#ifndef USE_ASYNC_COPY
    uint4 aRegH[2], aRegL[2];
#endif

    // ---- issue loads for step t into LDS buffer `buf` -----------------------
    auto issueStage = [&](int t, int buf) {
        int s = 0;
        #pragma unroll
        for (int i = 1; i < 5; ++i) s += (t >= scum[i]);
        const int kb   = (t - scum[s]) * 32;
        const int acol = aoff[s] + kb;
        cgfloat* B = bases[s];

        #pragma unroll
        for (int i = 0; i < 2; ++i) {
            size_t g = (size_t)(m0 + ar[i]) * KTOT + (size_t)(acol + akc[i]);
            int    l = buf * ASZ + ar[i] * LDA + akc[i];
#ifdef USE_ASYNC_COPY
            __builtin_amdgcn_global_load_async_to_lds_b128(GAS1(Ahi + g), LAS3(&AsH[l]), 0, 0);
            __builtin_amdgcn_global_load_async_to_lds_b128(GAS1(Alo + g), LAS3(&AsL[l]), 0, 0);
#else
            aRegH[i] = *(const uint4*)&Ahi[g];
            aRegL[i] = *(const uint4*)&Alo[g];
#endif
        }
        #pragma unroll
        for (int i = 0; i < 4; ++i)
            bReg[i] = *(cgv4f*)&B[(size_t)(kb + bk[i]) * 4096 + (size_t)(n0 + bcg[i])];
    };

    // ---- finish staging for buffer `buf`: bf16 split + LDS stores -----------
    auto storeStage = [&](int buf) {
#ifndef USE_ASYNC_COPY
        #pragma unroll
        for (int i = 0; i < 2; ++i) {
            int l = buf * ASZ + ar[i] * LDA + akc[i];
            *(uint4*)&AsH[l] = aRegH[i];
            *(uint4*)&AsL[l] = aRegL[i];
        }
#endif
        #pragma unroll
        for (int i = 0; i < 4; ++i) {
            v4f f = bReg[i];
            unsigned short h0 = f2bf(f[0]), h1 = f2bf(f[1]);
            unsigned short h2 = f2bf(f[2]), h3 = f2bf(f[3]);
            unsigned short l0 = f2bf(f[0] - bf2f(h0));
            unsigned short l1 = f2bf(f[1] - bf2f(h1));
            unsigned short l2 = f2bf(f[2] - bf2f(h2));
            unsigned short l3 = f2bf(f[3] - bf2f(h3));
            uint2 ph, pl;
            ph.x = (unsigned)h0 | ((unsigned)h1 << 16);
            ph.y = (unsigned)h2 | ((unsigned)h3 << 16);
            pl.x = (unsigned)l0 | ((unsigned)l1 << 16);
            pl.y = (unsigned)l2 | ((unsigned)l3 << 16);
            int l = buf * BSZ + bk[i] * LDB + bcg[i];
            *(uint2*)&BsH[l] = ph;
            *(uint2*)&BsL[l] = pl;
        }
    };

    // ---- 24 WMMAs on buffer `buf` -------------------------------------------
    auto doCompute = [&](int buf) {
        const int ab = buf * ASZ;
        const int bb = buf * BSZ;
        v16bf aH[4], aL[4];
        #pragma unroll
        for (int tm = 0; tm < 4; ++tm) {
            int r = waveM * 64 + tm * 16 + lm;
            FragU f;
            f.q[0] = *(const uint4*)&AsH[ab + r * LDA + lh * 8];
            f.q[1] = *(const uint4*)&AsH[ab + r * LDA + 16 + lh * 8];
            aH[tm] = f.v;
            f.q[0] = *(const uint4*)&AsL[ab + r * LDA + lh * 8];
            f.q[1] = *(const uint4*)&AsL[ab + r * LDA + 16 + lh * 8];
            aL[tm] = f.v;
        }
        #pragma unroll
        for (int tn = 0; tn < 2; ++tn) {
            int nn = waveN * 32 + tn * 16;
            FragU f, g;
            unsigned short* ph = (unsigned short*)f.q;
            unsigned short* pl = (unsigned short*)g.q;
            #pragma unroll
            for (int s = 0; s < 16; ++s) {
                int k = (s < 8) ? (lh * 8 + s) : (16 + lh * 8 + (s - 8));
                ph[s] = BsH[bb + k * LDB + nn + lm];
                pl[s] = BsL[bb + k * LDB + nn + lm];
            }
            v16bf bH = f.v;
            v16bf bL = g.v;
            #pragma unroll
            for (int tm = 0; tm < 4; ++tm) {
                // split precision: hi*hi + hi*lo + lo*hi (drop lo*lo)
                acc[tm][tn] = __builtin_amdgcn_wmma_f32_16x16x32_bf16(
                    false, aH[tm], false, bH, (short)0, acc[tm][tn], false, false);
                acc[tm][tn] = __builtin_amdgcn_wmma_f32_16x16x32_bf16(
                    false, aH[tm], false, bL, (short)0, acc[tm][tn], false, false);
                acc[tm][tn] = __builtin_amdgcn_wmma_f32_16x16x32_bf16(
                    false, aL[tm], false, bH, (short)0, acc[tm][tn], false, false);
            }
        }
    };

    // ---- software pipeline ---------------------------------------------------
    issueStage(0, 0);
    storeStage(0);
    for (int t = 0; t < NSTEP; ++t) {
        const int p = t & 1;
#ifdef USE_ASYNC_COPY
        __builtin_amdgcn_s_wait_asynccnt(0);
#endif
        __syncthreads();
        const bool hasNext = (t + 1) < NSTEP;
        if (hasNext) issueStage(t + 1, p ^ 1);   // global latency hides under WMMAs
        doCompute(p);
        if (hasNext) storeStage(p ^ 1);          // bf16 split + ds_store after compute
    }

    // ---- epilogue: C layout VGPR j -> M = j + 8*(lane>>4), N = lane&15 ------
    #pragma unroll
    for (int tm = 0; tm < 4; ++tm) {
        #pragma unroll
        for (int tn = 0; tn < 2; ++tn) {
            int mbase = m0 + waveM * 64 + tm * 16 + lh * 8;
            int nn    = n0 + waveN * 32 + tn * 16 + lm;
            #pragma unroll
            for (int j = 0; j < 8; ++j)
                out[(size_t)(mbase + j) * 4096 + nn] = acc[tm][tn][j];
        }
    }
}

// ---------------------------------------------------------------------------
extern "C" void kernel_launch(void* const* d_in, const int* in_sizes, int n_in,
                              void* d_out, int out_size, void* d_ws, size_t ws_size,
                              hipStream_t stream) {
    (void)in_sizes; (void)n_in; (void)out_size; (void)ws_size;

    const float* x  = (const float*)d_in[0];
    const float* b0 = (const float*)d_in[1];
    const float* b1 = (const float*)d_in[2];
    const float* b2 = (const float*)d_in[3];
    const float* b3 = (const float*)d_in[4];
    const float* b4 = (const float*)d_in[5];
    float* out = (float*)d_out;

    // workspace: bf16 hi plane then bf16 lo plane, each [4096][KTOT]
    unsigned short* Ahi = (unsigned short*)d_ws;
    unsigned short* Alo = Ahi + (size_t)4096 * KTOT;

    dwt_db4_kernel<<<4096, 256, 0, stream>>>(x, Ahi, Alo);

    dim3 grid(32, 32);  // N tiles x M tiles
    wkan_wmma_gemm<<<grid, 256, 0, stream>>>(Ahi, Alo, b0, b1, b2, b3, b4, out);
}
